// Wavenet_39513699123421
// MI455X (gfx1250) — matrix operand order repaired and checked by reading, not compiled, not run on hardware
//
#include <hip/hip_runtime.h>

// ---------------------------------------------------------------------------
// WaveNet forward, fused per-layer bf16-WMMA kernels for gfx1250 (MI455X).
//   B=2, T=16384, L=16, R=128, S=S2=O=256, C=128, V=256, UP=256, GAIN=1.
// FLOPs ~129G; working set (~55MB: x ping-pong bf16, skip fp32, weights)
// stays resident in the 192MB L2, so per-layer global round-trips are cheap.
// All GEMMs go through v_wmma_f32_16x16x32_bf16 (wave32, 16x16 tiles).
// ---------------------------------------------------------------------------

typedef __attribute__((ext_vector_type(16))) __bf16          v16bf;
typedef __attribute__((ext_vector_type(8)))  float           v8f;
typedef __attribute__((ext_vector_type(8)))  unsigned short  v8u16;
typedef __attribute__((ext_vector_type(16))) unsigned short  v16u16;

#define WN_T   16384
#define WN_B   2
#define WN_R   128
#define WN_S   256
#define WN_NT  64      // time-tile width per workgroup

static __device__ __forceinline__ unsigned short f2b(float f) {
    unsigned u = __builtin_bit_cast(unsigned, f);
    unsigned r = u + 0x7FFFu + ((u >> 16) & 1u);   // round-to-nearest-even
    return (unsigned short)(r >> 16);
}
static __device__ __forceinline__ float b2f(unsigned short h) {
    unsigned u = ((unsigned)h) << 16;
    return __builtin_bit_cast(float, u);
}
static __device__ __forceinline__ v8f vzero8() {
    v8f z = {0.f, 0.f, 0.f, 0.f, 0.f, 0.f, 0.f, 0.f};
    return z;
}
static __device__ __forceinline__ v8f wmma_bf16(v16bf a, v16bf b, v8f c) {
    // D = A(16x32 bf16) x B(32x16 bf16) + C(16x16 f32)
    return __builtin_amdgcn_wmma_f32_16x16x32_bf16(
        false, a, false, b, (short)0, c, false, false);
}

// A fragment (16x32 bf16) from LDS stored [M][K] row-major, row stride 128.
// Per ISA: lane L holds M=L%16; halves 0..7 -> K = base..base+7,
//          halves 8..15 -> K = base+16..base+23, base = kstep*32 + (L/16)*8.
static __device__ __forceinline__ v16bf ldsA(const unsigned short* W,
                                             int mbase, int lane, int kstep) {
    int m  = mbase + (lane & 15);
    int kb = kstep * 32 + ((lane >> 4) << 3);
    const v8u16 lo = *(const v8u16*)(W + m * 128 + kb);
    const v8u16 hi = *(const v8u16*)(W + m * 128 + kb + 16);
    v16u16 c = __builtin_shufflevector(lo, hi, 0,1,2,3,4,5,6,7,
                                               8,9,10,11,12,13,14,15);
    return __builtin_bit_cast(v16bf, c);
}

// B fragment (32x16 bf16) from LDS stored [N(t)][K] row-major, row stride `str`.
// Per ISA: lane L holds N=L%16; halves 0..7 -> K = base..base+7,
//          halves 8..15 -> K = base+16..base+23, base = kstep*32 + (L/16)*8.
static __device__ __forceinline__ v16bf ldsB(const unsigned short* X,
                                             int nbase, int lane, int kstep,
                                             int str) {
    int n  = nbase + (lane & 15);
    int kb = kstep * 32 + ((lane >> 4) << 3);
    const v8u16 lo = *(const v8u16*)(X + n * str + kb);
    const v8u16 hi = *(const v8u16*)(X + n * str + kb + 16);
    v16u16 c = __builtin_shufflevector(lo, hi, 0,1,2,3,4,5,6,7,
                                               8,9,10,11,12,13,14,15);
    return __builtin_bit_cast(v16bf, c);
}

// ---------------------------------------------------------------------------
// cond_small[b][o][frame] = cond_W @ features + cond_b  (cond is constant over
// each 256-sample frame -> only B*256*64 = 32768 distinct values)
// ---------------------------------------------------------------------------
__global__ __launch_bounds__(256)
void wn_cond(const float* __restrict__ feat, const float* __restrict__ cW,
             const float* __restrict__ cb, float* __restrict__ conds) {
    int gid   = blockIdx.x * 256 + threadIdx.x;   // 32768 total
    int frame = gid & 63;
    int o     = (gid >> 6) & 255;
    int b     = gid >> 14;
    float acc = cb[o];
    #pragma unroll 4
    for (int c = 0; c < 128; ++c)
        acc += cW[o * 128 + c] * feat[(b * 128 + c) * 64 + frame];
    conds[gid] = acc;
}

// ---------------------------------------------------------------------------
// x0[b][r][t] = bf16(embed[audio[b][t]][r])
// ---------------------------------------------------------------------------
__global__ __launch_bounds__(256)
void wn_embed(const int* __restrict__ audio, const float* __restrict__ embed,
              unsigned short* __restrict__ x0) {
    int blk    = blockIdx.x;          // (b*128 + r) * 64 + tchunk
    int rIdx   = blk >> 6;            // b*128 + r
    int t      = (blk & 63) * 256 + threadIdx.x;
    int b      = rIdx >> 7;
    int r      = rIdx & 127;
    int a      = audio[b * WN_T + t];
    x0[(size_t)rIdx * WN_T + t] = f2b(embed[a * 128 + r]);
}

// ---------------------------------------------------------------------------
// One residual block. 256 threads = 8 waves; each workgroup owns 64 columns.
// LDS (u16): Wb[256*128] | xc[64*128] | xp[64*128] | at[64*128]  = 112 KB
// ---------------------------------------------------------------------------
__global__ __launch_bounds__(256)
void wn_layer(const unsigned short* __restrict__ xin,
              unsigned short* __restrict__ xout,
              float* __restrict__ skip, const float* __restrict__ conds,
              const float* __restrict__ dW, const float* __restrict__ db,
              const float* __restrict__ sW, const float* __restrict__ sb,
              const float* __restrict__ rW, const float* __restrict__ rb,
              int li, int d, int isFirst, int hasRes) {
    extern __shared__ unsigned short smem[];
    unsigned short* Wb = smem;                 // 256x128 weight panel (bf16)
    unsigned short* xc = smem + 32768;         // x tile   [t][r]
    unsigned short* xp = xc + 8192;            // x[t-d]   [t][r]
    unsigned short* at = xp + 8192;            // acts     [t][r]

    const int tid  = threadIdx.x;
    const int lane = tid & 31;
    const int wave = tid >> 5;
    const int tile = blockIdx.x & 255;
    const int b    = blockIdx.x >> 8;
    const int t0   = tile * WN_NT;

    // Warm next weight panel in L2 while we compute the dilate taps.
    __builtin_prefetch(sW + (size_t)li * 32768, 0, 1);

    // ---- stage x tiles (bf16 passthrough), transposed to [t][r] ----
    for (int e = tid; e < 8192; e += 256) {
        int r = e >> 6, j = e & 63;
        xc[j * 128 + r] = xin[(size_t)(b * 128 + r) * WN_T + (t0 + j)];
        int tp = t0 + j - d;
        xp[j * 128 + r] = (tp >= 0)
            ? xin[(size_t)(b * 128 + r) * WN_T + tp] : (unsigned short)0;
    }

    // ---- dilate GEMMs: in_act = W1 @ x + W0 @ x[t-d]  (256x128 x 128x64) ----
    v8f acc[2][4];
    #pragma unroll
    for (int m = 0; m < 2; ++m)
        #pragma unroll
        for (int n = 0; n < 4; ++n) acc[m][n] = vzero8();

    #pragma unroll
    for (int tap = 0; tap < 2; ++tap) {
        __syncthreads();                              // Wb free / x staged
        const int tapSel = (tap == 0) ? 1 : 0;        // current tap first
        const float* ws  = dW + (size_t)li * 65536 + tapSel;
        for (int e = tid; e < 32768; e += 256)
            Wb[e] = f2b(ws[(size_t)e * 2]);           // (L,2R,R,2) layout
        __syncthreads();
        const unsigned short* Bt = (tap == 0) ? xc : xp;
        #pragma unroll
        for (int k = 0; k < 4; ++k) {
            v16bf a0 = ldsA(Wb, 16 * wave,       lane, k);
            v16bf a1 = ldsA(Wb, 16 * (wave + 8), lane, k);
            #pragma unroll
            for (int nt = 0; nt < 4; ++nt) {
                v16bf bf = ldsB(Bt, nt * 16, lane, k, 128);
                acc[0][nt] = wmma_bf16(a0, bf, acc[0][nt]);
                acc[1][nt] = wmma_bf16(a1, bf, acc[1][nt]);
            }
        }
    }

    // ---- gating in registers: wave w owns rows 16w.. (tanh) and 128+16w.. ----
    const int frame = t0 >> 8;
    const int mrow  = (lane >> 4) << 3;
    const int ncol  = lane & 15;
    #pragma unroll
    for (int nt = 0; nt < 4; ++nt) {
        #pragma unroll
        for (int v = 0; v < 8; ++v) {
            int ot = 16 * wave + mrow + v;
            int os = 128 + ot;
            float it = acc[0][nt][v] + db[li * 256 + ot]
                     + conds[(size_t)(b * 256 + ot) * 64 + frame];
            float is = acc[1][nt][v] + db[li * 256 + os]
                     + conds[(size_t)(b * 256 + os) * 64 + frame];
            float g = tanhf(it) * (1.0f / (1.0f + __expf(-is)));
            at[(nt * 16 + ncol) * 128 + (16 * wave + mrow + v)] = f2b(g);
        }
    }
    __syncthreads();   // acts visible; dilate weight reads retired

    // ---- skip GEMM: skip += skip_W @ acts + skip_b ----
    for (int e = tid; e < 32768; e += 256)
        Wb[e] = f2b(sW[(size_t)li * 32768 + e]);
    __syncthreads();
    v8f as[2][4];
    #pragma unroll
    for (int m = 0; m < 2; ++m)
        #pragma unroll
        for (int n = 0; n < 4; ++n) as[m][n] = vzero8();
    #pragma unroll
    for (int k = 0; k < 4; ++k) {
        v16bf a0 = ldsA(Wb, 16 * wave,       lane, k);
        v16bf a1 = ldsA(Wb, 16 * (wave + 8), lane, k);
        #pragma unroll
        for (int nt = 0; nt < 4; ++nt) {
            v16bf bf = ldsB(at, nt * 16, lane, k, 128);
            as[0][nt] = wmma_bf16(a0, bf, as[0][nt]);
            as[1][nt] = wmma_bf16(a1, bf, as[1][nt]);
        }
    }
    #pragma unroll
    for (int m = 0; m < 2; ++m)
        #pragma unroll
        for (int nt = 0; nt < 4; ++nt)
            #pragma unroll
            for (int v = 0; v < 8; ++v) {
                int s = 16 * (wave + 8 * m) + mrow + v;
                int t = t0 + nt * 16 + ncol;
                size_t ad = (size_t)(b * 256 + s) * WN_T + t;
                float val = as[m][nt][v] + sb[li * 256 + s];
                if (isFirst) skip[ad] = val;
                else         skip[ad] += val;
            }

    // ---- res GEMM + residual: x_out = res_W @ acts + res_b + x ----
    if (hasRes) {
        __syncthreads();
        for (int e = tid; e < 16384; e += 256)
            Wb[e] = f2b(rW[(size_t)li * 16384 + e]);
        __syncthreads();
        v8f ar[4];
        #pragma unroll
        for (int n = 0; n < 4; ++n) ar[n] = vzero8();
        #pragma unroll
        for (int k = 0; k < 4; ++k) {
            v16bf a0 = ldsA(Wb, 16 * wave, lane, k);   // 8 m-tiles, 1/wave
            #pragma unroll
            for (int nt = 0; nt < 4; ++nt) {
                v16bf bf = ldsB(at, nt * 16, lane, k, 128);
                ar[nt] = wmma_bf16(a0, bf, ar[nt]);
            }
        }
        #pragma unroll
        for (int nt = 0; nt < 4; ++nt)
            #pragma unroll
            for (int v = 0; v < 8; ++v) {
                int r    = 16 * wave + mrow + v;
                int tloc = nt * 16 + ncol;
                float xo = b2f(xc[tloc * 128 + r]);    // x tile still live
                float xn = ar[nt][v] + rb[(size_t)li * 128 + r] + xo;
                xout[(size_t)(b * 128 + r) * WN_T + (t0 + tloc)] = f2b(xn);
            }
    }
}

// ---------------------------------------------------------------------------
// Epilogue: y = end_W @ relu(out_W @ relu(skip)), shifted right by 1 sample.
// LDS (u16): Wb[256*128] | aT[64*256]  = 96 KB
// ---------------------------------------------------------------------------
__global__ __launch_bounds__(256)
void wn_out(const float* __restrict__ skip, const float* __restrict__ oW,
            const float* __restrict__ eW, float* __restrict__ out) {
    extern __shared__ unsigned short smem[];
    unsigned short* Wb = smem;                // 256x128 weight half-panel
    unsigned short* aT = smem + 32768;        // activations [t][c], c=256

    const int tid  = threadIdx.x;
    const int lane = tid & 31;
    const int wave = tid >> 5;
    const int tile = blockIdx.x & 255;
    const int b    = blockIdx.x >> 8;
    const int t0   = tile * WN_NT;
    const int mrow = (lane >> 4) << 3;
    const int ncol = lane & 15;

    // relu(skip) -> [t][s] bf16
    for (int e = tid; e < 16384; e += 256) {
        int s = e >> 6, j = e & 63;
        float f = skip[(size_t)(b * 256 + s) * WN_T + (t0 + j)];
        aT[j * 256 + s] = f2b(fmaxf(f, 0.f));
    }

    // GEMM1: h = out_W @ relu(skip), K=256 in two staged 128-halves
    v8f ah[2][4];
    #pragma unroll
    for (int m = 0; m < 2; ++m)
        #pragma unroll
        for (int n = 0; n < 4; ++n) ah[m][n] = vzero8();
    #pragma unroll
    for (int kh = 0; kh < 2; ++kh) {
        __syncthreads();
        for (int e = tid; e < 32768; e += 256) {
            int o = e >> 7, r = e & 127;
            Wb[e] = f2b(oW[o * 256 + kh * 128 + r]);
        }
        __syncthreads();
        #pragma unroll
        for (int k = 0; k < 4; ++k) {
            v16bf a0 = ldsA(Wb, 16 * wave,       lane, k);
            v16bf a1 = ldsA(Wb, 16 * (wave + 8), lane, k);
            #pragma unroll
            for (int nt = 0; nt < 4; ++nt) {
                v16bf bf = ldsB(aT, nt * 16, lane, kh * 4 + k, 256);
                ah[0][nt] = wmma_bf16(a0, bf, ah[0][nt]);
                ah[1][nt] = wmma_bf16(a1, bf, ah[1][nt]);
            }
        }
    }
    __syncthreads();   // everyone done reading aT
    #pragma unroll
    for (int m = 0; m < 2; ++m)
        #pragma unroll
        for (int nt = 0; nt < 4; ++nt)
            #pragma unroll
            for (int v = 0; v < 8; ++v) {
                int o = 16 * (wave + 8 * m) + mrow + v;
                aT[(nt * 16 + ncol) * 256 + o] = f2b(fmaxf(ah[m][nt][v], 0.f));
            }

    // GEMM2: y = end_W @ relu(h)
    v8f ay[2][4];
    #pragma unroll
    for (int m = 0; m < 2; ++m)
        #pragma unroll
        for (int n = 0; n < 4; ++n) ay[m][n] = vzero8();
    #pragma unroll
    for (int kh = 0; kh < 2; ++kh) {
        __syncthreads();
        for (int e = tid; e < 32768; e += 256) {
            int o = e >> 7, r = e & 127;
            Wb[e] = f2b(eW[o * 256 + kh * 128 + r]);
        }
        __syncthreads();
        #pragma unroll
        for (int k = 0; k < 4; ++k) {
            v16bf a0 = ldsA(Wb, 16 * wave,       lane, k);
            v16bf a1 = ldsA(Wb, 16 * (wave + 8), lane, k);
            #pragma unroll
            for (int nt = 0; nt < 4; ++nt) {
                v16bf bf = ldsB(aT, nt * 16, lane, kh * 4 + k, 256);
                ay[0][nt] = wmma_bf16(a0, bf, ay[0][nt]);
                ay[1][nt] = wmma_bf16(a1, bf, ay[1][nt]);
            }
        }
    }

    // store with +1 shift (out[:, :, 0] = 0, out[:, :, T-1] input dropped)
    #pragma unroll
    for (int m = 0; m < 2; ++m)
        #pragma unroll
        for (int nt = 0; nt < 4; ++nt)
            #pragma unroll
            for (int v = 0; v < 8; ++v) {
                int o = 16 * (wave + 8 * m) + mrow + v;
                int t = t0 + nt * 16 + ncol;
                size_t rowb = (size_t)(b * 256 + o) * WN_T;
                if (t + 1 < WN_T) out[rowb + t + 1] = ay[m][nt][v];
                if (t == 0)       out[rowb] = 0.f;
            }
}

// ---------------------------------------------------------------------------
extern "C" void kernel_launch(void* const* d_in, const int* in_sizes, int n_in,
                              void* d_out, int out_size, void* d_ws,
                              size_t ws_size, hipStream_t stream) {
    const float* features = (const float*)d_in[0];
    const int*   audio    = (const int*)d_in[1];
    const float* embed    = (const float*)d_in[2];
    const float* cond_W   = (const float*)d_in[3];
    const float* cond_b   = (const float*)d_in[4];
    const float* dilate_W = (const float*)d_in[5];
    const float* dilate_b = (const float*)d_in[6];
    const float* res_W    = (const float*)d_in[7];
    const float* res_b    = (const float*)d_in[8];
    const float* skip_W   = (const float*)d_in[9];
    const float* skip_b   = (const float*)d_in[10];
    const float* out_W    = (const float*)d_in[11];
    const float* end_W    = (const float*)d_in[12];
    float* out = (float*)d_out;

    // Workspace: x ping-pong (bf16), skip (fp32), cond_small (fp32) ~= 48.2MB
    const size_t XN = (size_t)WN_B * WN_R * WN_T;        // 4,194,304
    unsigned short* x0 = (unsigned short*)d_ws;
    unsigned short* x1 = x0 + XN;
    float* skip  = (float*)(x1 + XN);
    float* conds = skip + (size_t)WN_B * WN_S * WN_T;

    static const int DIL[16] = {1, 2, 4, 8, 16, 32, 64, 128, 256,
                                1, 2, 4, 8, 16, 32, 64};

    wn_cond<<<128, 256, 0, stream>>>(features, cond_W, cond_b, conds);
    wn_embed<<<16384, 256, 0, stream>>>(audio, embed, x0);

    const int    grid      = WN_B * (WN_T / WN_NT);      // 512 workgroups
    const size_t layerSmem = (size_t)(32768 + 3 * 8192) * 2;  // 112 KB
    for (int i = 0; i < 16; ++i) {
        unsigned short* xin  = (i & 1) ? x1 : x0;
        unsigned short* xout = (i & 1) ? x0 : x1;
        wn_layer<<<grid, 256, layerSmem, stream>>>(
            xin, xout, skip, conds, dilate_W, dilate_b, skip_W, skip_b,
            res_W, res_b, i, DIL[i], (i == 0) ? 1 : 0, (i < 15) ? 1 : 0);
    }

    const size_t outSmem = (size_t)(32768 + 16384) * 2;  // 96 KB
    wn_out<<<grid, 256, outSmem, stream>>>(skip, out_W, end_W, out);
}
